// Attention_82351702934023
// MI455X (gfx1250) — matrix operand order repaired and checked by reading, not compiled, not run on hardware
//
#include <hip/hip_runtime.h>
#include <hip/hip_bf16.h>

// ---------------------------------------------------------------------------
// CDNA5 (gfx1250) causal multi-head attention, bf16 WMMA with f32 accumulate.
// B=4, S=2048, D=1024, H=16, DK=DV=64.
// All LDS tile staging via the Tensor Data Mover, double-buffered so the DMA
// overlaps WMMA compute.  Weights are pre-transposed at conversion time so no
// in-kernel transposition is needed anywhere.
// ---------------------------------------------------------------------------

typedef __attribute__((ext_vector_type(16))) __bf16 bf16x16;
typedef __attribute__((ext_vector_type(8)))  __bf16 bf16x8;
typedef __attribute__((ext_vector_type(8)))  float  f32x8;
typedef __attribute__((ext_vector_type(4)))  unsigned int u32x4;
typedef __attribute__((ext_vector_type(8)))  int i32x8;
typedef __attribute__((ext_vector_type(4)))  int i32x4;

union AFrag {
    bf16x16 v;
    bf16x8  h[2];
};

#define WMMA_BF16(a, b, c) \
    __builtin_amdgcn_wmma_f32_16x16x32_bf16(false, (a), false, (b), (short)0, (c), false, false)

#define B_  4
#define S_  2048
#define D_  1024
#define H_  16
#define DK_ 64
#define DV_ 64

// --------------------------- TDM support -----------------------------------
#if defined(__has_builtin)
#if __has_builtin(__builtin_amdgcn_tensor_load_to_lds) && __has_builtin(__builtin_amdgcn_s_wait_tensorcnt)
#define HAVE_TDM 1
#endif
#endif

#if defined(HAVE_TDM)
#if __has_include(<hip/amd_detail/amd_gfx1250_TDM.h>)
#define TDM_6ARG 1
#endif

// Issue a TDM 2D tile load (bf16 elements) into LDS.  Descriptor per ISA ch.8.
// tile_d0 = contiguous elements per row, tile_d1 = rows, stride0 = row stride.
__device__ __forceinline__ void tdm_load_2d(unsigned lds_off, const void* gptr,
                                            unsigned tile_d0, unsigned tile_d1,
                                            unsigned long long stride0)
{
    unsigned long long ga = (unsigned long long)gptr;
    u32x4 g0;
    g0[0] = 1u;                                    // count=1, user mode
    g0[1] = lds_off;                               // lds_addr (bytes)
    g0[2] = (unsigned)(ga & 0xffffffffu);          // global_addr lo
    g0[3] = (unsigned)((ga >> 32) & 0x01ffffffu)   // global_addr hi
          | 0x80000000u;                           // type=2
    i32x8 g1;
    g1[0] = (int)(1u << 16);                       // data_size=1 (2 bytes)
    g1[1] = (int)((tile_d0 & 0xffffu) << 16);      // tensor_dim0 = tile_d0
    g1[2] = (int)((tile_d1 & 0xffffu) << 16);      // tensor_dim1 = tile_d1
    g1[3] = (int)((tile_d0 & 0xffffu) << 16);      // tile_dim0
    g1[4] = (int)(tile_d1 & 0xffffu);              // tile_dim1 (tile_dim2 = 0)
    g1[5] = (int)(stride0 & 0xffffffffu);          // tensor_dim0_stride lo
    g1[6] = (int)((stride0 >> 32) & 0xffffu);      // tensor_dim0_stride hi
    g1[7] = 0;
    i32x4 z4 = {};
#if defined(TDM_6ARG)
    i32x8 z8 = {};
    __builtin_amdgcn_tensor_load_to_lds(g0, g1, z4, z4, z8, 0);
#else
    __builtin_amdgcn_tensor_load_to_lds(g0, g1, z4, z4, 0);
#endif
}
#define TDM_WAIT() __builtin_amdgcn_s_wait_tensorcnt(0)
#endif  // HAVE_TDM

__device__ __forceinline__ unsigned lds_offset(const void* p) {
    return (unsigned)(unsigned long long)p;   // AS(3) flat addr: low 32 bits = LDS offset
}

// ---------------------------------------------------------------------------
// Kernel 0a: f32 -> bf16 conversion (x)
// ---------------------------------------------------------------------------
__global__ void cvt_f32_bf16(const float* __restrict__ src, __bf16* __restrict__ dst, int n) {
    int i = blockIdx.x * blockDim.x + threadIdx.x;
    if (i < n) dst[i] = (__bf16)src[i];
}

// ---------------------------------------------------------------------------
// Kernel 0b: f32 [G, R, C] -> bf16 transposed [G, C, R]   (weights)
// ---------------------------------------------------------------------------
__global__ void cvt_transpose_bf16(const float* __restrict__ src, __bf16* __restrict__ dst,
                                   int R, int C) {
    size_t base = (size_t)blockIdx.z * R * C;
    int i = blockIdx.x * blockDim.x + threadIdx.x;
    if (i < R * C) {
        int r = i / C, c = i % C;
        dst[base + (size_t)c * R + r] = (__bf16)src[base + i];
    }
}

// ---------------------------------------------------------------------------
// Kernel 1: QKV projection.  grid = (S/128, 3*H, B), block = 256 (8 waves).
// Weights pre-transposed: wt [H, 64(n), D(k)].  k-step = 64, double-buffered.
// Q, K stored [B,H,S,DK];  V stored transposed [B,H,DV,S].
// ---------------------------------------------------------------------------
__global__ void __launch_bounds__(256)
qkv_gemm(const __bf16* __restrict__ xb,
         const __bf16* __restrict__ wqt,
         const __bf16* __restrict__ wkt,
         const __bf16* __restrict__ wvt,
         __bf16* __restrict__ Qo,
         __bf16* __restrict__ Ko,
         __bf16* __restrict__ Vt)
{
    __shared__ __bf16 Xs[2][128 * 64];   // [row][k]
    __shared__ __bf16 Ws[2][64 * 64];    // [n][k]

    const int b    = blockIdx.z;
    const int proj = blockIdx.y / H_;
    const int h    = blockIdx.y % H_;
    const int s0   = blockIdx.x * 128;

    const __bf16* wt = ((proj == 0) ? wqt : (proj == 1) ? wkt : wvt)
                     + (size_t)h * DK_ * D_;
    const __bf16* xtile0 = xb + ((size_t)b * S_ + s0) * D_;

    const int tid  = threadIdx.x;
    const int lane = tid & 31;
    const int wid  = tid >> 5;
    const float scl = (proj == 0) ? 0.125f : 1.0f;   // 1/sqrt(DK) folded into Q

    f32x8 acc[4] = {};

#if defined(HAVE_TDM)
    if (tid < 32) {
        tdm_load_2d(lds_offset(Xs[0]), xtile0, 64, 128, (unsigned long long)D_);
        tdm_load_2d(lds_offset(Ws[0]), wt,     64, 64,  (unsigned long long)D_);
    }
#endif

    for (int k0 = 0; k0 < D_; k0 += 64) {
        const int cur = (k0 >> 6) & 1;
#if defined(HAVE_TDM)
        if (tid < 32) TDM_WAIT();
        __syncthreads();
        if (k0 + 64 < D_ && tid < 32) {   // prefetch next tiles into other buffer
            tdm_load_2d(lds_offset(Xs[cur ^ 1]), xtile0 + k0 + 64, 64, 128,
                        (unsigned long long)D_);
            tdm_load_2d(lds_offset(Ws[cur ^ 1]), wt + k0 + 64, 64, 64,
                        (unsigned long long)D_);
        }
#else
        {
            const uint4* xg4 = (const uint4*)(xtile0 + k0);
            uint4* xs4 = (uint4*)Xs[cur];
#pragma unroll
            for (int t = 0; t < 4; ++t) {
                int i = tid + t * 256;
                int r = i >> 3, c = i & 7;           // 8 uint4 per 64-elem row
                xs4[r * 8 + c] = xg4[(size_t)r * (D_ / 8) + c];
            }
            const uint4* wg4 = (const uint4*)(wt + k0);
            uint4* ws4 = (uint4*)Ws[cur];
#pragma unroll
            for (int t = 0; t < 2; ++t) {
                int i = tid + t * 256;
                int r = i >> 3, c = i & 7;
                ws4[r * 8 + c] = wg4[(size_t)r * (D_ / 8) + c];
            }
        }
        __syncthreads();
#endif
        const __bf16* Xc = Xs[cur];
        const __bf16* Wc = Ws[cur];
#pragma unroll
        for (int ks = 0; ks < 2; ++ks) {
            AFrag a;
            {
                int m  = lane & 15;
                int kb = (lane >> 4) * 8;
                const __bf16* base = Xc + (wid * 16 + m) * 64 + ks * 32;
                a.h[0] = *(const bf16x8*)(base + kb);
                a.h[1] = *(const bf16x8*)(base + 16 + kb);
            }
#pragma unroll
            for (int nt = 0; nt < 4; ++nt) {
                AFrag bf;
                int n  = nt * 16 + (lane & 15);
                int kb = (lane >> 4) * 8;
                const __bf16* base = Wc + n * 64 + ks * 32 + kb;
                bf.h[0] = *(const bf16x8*)(base);
                bf.h[1] = *(const bf16x8*)(base + 16);
                acc[nt] = WMMA_BF16(a.v, bf.v, acc[nt]);
            }
        }
        __syncthreads();
    }

    // C layout: n = lane&15, m = (lane>>4)*8 + r
    const int n0 = lane & 15;
    const int m0 = (lane >> 4) * 8;
    if (proj == 2) {
        __bf16* out = Vt + ((size_t)b * H_ + h) * DV_ * S_;   // [dv][s]
#pragma unroll
        for (int nt = 0; nt < 4; ++nt)
#pragma unroll
            for (int r = 0; r < 8; ++r) {
                int s = s0 + wid * 16 + m0 + r;
                out[(size_t)(nt * 16 + n0) * S_ + s] = (__bf16)acc[nt][r];
            }
    } else {
        __bf16* out = ((proj == 0) ? Qo : Ko) + ((size_t)b * H_ + h) * S_ * DK_;
#pragma unroll
        for (int nt = 0; nt < 4; ++nt)
#pragma unroll
            for (int r = 0; r < 8; ++r) {
                int s = s0 + wid * 16 + m0 + r;
                out[(size_t)s * DK_ + nt * 16 + n0] = (__bf16)(acc[nt][r] * scl);
            }
    }
}

// ---------------------------------------------------------------------------
// Kernel 2: causal flash attention.  grid = (S/64, H, B), block = 128 (4 waves).
// Q,K: [B,H,S,DK]; V: transposed [B,H,DV,S]; O: heads [B,S,H*DV].
// K/V tiles double-buffered via TDM.
// ---------------------------------------------------------------------------
__global__ void __launch_bounds__(128)
flash_attn(const __bf16* __restrict__ Q,
           const __bf16* __restrict__ K,
           const __bf16* __restrict__ Vt,
           __bf16* __restrict__ O)
{
    __shared__ __bf16 Ks[2][64 * 64];   // [key][d]
    __shared__ __bf16 Vs[2][64 * 64];   // [dv][key]
    __shared__ __bf16 Ps[4][16 * 64];   // per wave: [m][key]

    const int b  = blockIdx.z;
    const int h  = blockIdx.y;
    const int qt = blockIdx.x;
    const int q0 = qt * 64;

    const int tid  = threadIdx.x;
    const int lane = tid & 31;
    const int w    = tid >> 5;

    const size_t bh   = ((size_t)b * H_ + h) * S_;
    const __bf16* Kb  = K  + bh * DK_;
    const __bf16* Vtb = Vt + ((size_t)b * H_ + h) * DV_ * S_;

    // Preload Q fragments (2 k-steps of 32 over DK=64)
    AFrag qa[2];
    {
        int m  = lane & 15;
        int kb = (lane >> 4) * 8;
        const __bf16* qrow = Q + (bh + q0 + w * 16 + m) * DK_;
#pragma unroll
        for (int ks = 0; ks < 2; ++ks) {
            qa[ks].h[0] = *(const bf16x8*)(qrow + ks * 32 + kb);
            qa[ks].h[1] = *(const bf16x8*)(qrow + ks * 32 + 16 + kb);
        }
    }

    f32x8 acc[4] = {};
    float m_i[8], l_i[8];
#pragma unroll
    for (int r = 0; r < 8; ++r) { m_i[r] = -3.0e38f; l_i[r] = 0.0f; }

    const int m0 = (lane >> 4) * 8;
    const int n0 = lane & 15;

#if defined(HAVE_TDM)
    if (tid < 32) {
        tdm_load_2d(lds_offset(Ks[0]), Kb, 64, 64, 64ull);
        tdm_load_2d(lds_offset(Vs[0]), Vtb, 64, 64, (unsigned long long)S_);
    }
#endif

    for (int jt = 0; jt <= qt; ++jt) {
        const int cur = jt & 1;
#if defined(HAVE_TDM)
        if (tid < 32) TDM_WAIT();
        __syncthreads();
        if (jt + 1 <= qt && tid < 32) {   // prefetch next K/V tile during compute
            tdm_load_2d(lds_offset(Ks[cur ^ 1]), Kb + (size_t)(jt + 1) * 64 * DK_,
                        64, 64, 64ull);
            tdm_load_2d(lds_offset(Vs[cur ^ 1]), Vtb + (size_t)(jt + 1) * 64,
                        64, 64, (unsigned long long)S_);
        }
#else
        {
            const uint4* kg4 = (const uint4*)(Kb + (size_t)jt * 64 * DK_);
            uint4* ks4 = (uint4*)Ks[cur];
#pragma unroll
            for (int t = 0; t < 4; ++t)
                ks4[tid + t * 128] = kg4[tid + t * 128];      // contiguous 8KB

            uint4* vs4 = (uint4*)Vs[cur];
#pragma unroll
            for (int t = 0; t < 4; ++t) {
                int i = tid + t * 128;
                int dv = i >> 3, c = i & 7;                   // 8 uint4 per 64-key row
                vs4[dv * 8 + c] =
                    *(const uint4*)(Vtb + (size_t)dv * S_ + (size_t)jt * 64 + c * 8);
            }
        }
        __syncthreads();
#endif
        const __bf16* Kc = Ks[cur];
        const __bf16* Vc = Vs[cur];

        // ---- scores: S = Q (16x64) . K^T ---------------------------------
        f32x8 sc[4] = {};
#pragma unroll
        for (int nt = 0; nt < 4; ++nt) {
#pragma unroll
            for (int ks = 0; ks < 2; ++ks) {
                AFrag kf;
                int key = nt * 16 + n0;
                int db  = ks * 32 + (lane >> 4) * 8;
                const __bf16* base = Kc + key * 64 + db;
                kf.h[0] = *(const bf16x8*)(base);
                kf.h[1] = *(const bf16x8*)(base + 16);
                sc[nt] = WMMA_BF16(qa[ks].v, kf.v, sc[nt]);
            }
        }

        // ---- causal mask on diagonal tile --------------------------------
        if (jt == qt) {
#pragma unroll
            for (int nt = 0; nt < 4; ++nt) {
                int kidx = jt * 64 + nt * 16 + n0;
#pragma unroll
                for (int r = 0; r < 8; ++r) {
                    int qidx = q0 + w * 16 + m0 + r;
                    if (kidx > qidx) sc[nt][r] = -3.0e38f;
                }
            }
        }

        // ---- online softmax ----------------------------------------------
        float rmax[8];
#pragma unroll
        for (int r = 0; r < 8; ++r) {
            float mx = fmaxf(fmaxf(sc[0][r], sc[1][r]), fmaxf(sc[2][r], sc[3][r]));
#pragma unroll
            for (int d = 1; d < 16; d <<= 1)
                mx = fmaxf(mx, __shfl_xor(mx, d, 32));   // row lives in a 16-lane half
            rmax[r] = mx;
        }

        float alpha[8], rsum[8];
#pragma unroll
        for (int r = 0; r < 8; ++r) {
            float mnew = fmaxf(m_i[r], rmax[r]);
            alpha[r]   = __expf(m_i[r] - mnew);
            m_i[r]     = mnew;
            rsum[r]    = 0.0f;
        }

#pragma unroll
        for (int nt = 0; nt < 4; ++nt) {
#pragma unroll
            for (int r = 0; r < 8; ++r) {
                float p = __expf(sc[nt][r] - m_i[r]);
                rsum[r] += p;
                Ps[w][(m0 + r) * 64 + nt * 16 + n0] = (__bf16)p;
            }
        }
#pragma unroll
        for (int r = 0; r < 8; ++r) {
            float s = rsum[r];
#pragma unroll
            for (int d = 1; d < 16; d <<= 1)
                s += __shfl_xor(s, d, 32);
            l_i[r] = l_i[r] * alpha[r] + s;
        }

#pragma unroll
        for (int nt = 0; nt < 4; ++nt)
#pragma unroll
            for (int r = 0; r < 8; ++r)
                acc[nt][r] *= alpha[r];

        __syncthreads();   // Ps visible; tiles stable until PV done

        // ---- O += P (16x64keys) . V --------------------------------------
#pragma unroll
        for (int ntv = 0; ntv < 4; ++ntv) {
#pragma unroll
            for (int ks = 0; ks < 2; ++ks) {
                AFrag pa;
                {
                    int m  = lane & 15;
                    int kb = (lane >> 4) * 8;
                    const __bf16* base = &Ps[w][m * 64 + ks * 32];
                    pa.h[0] = *(const bf16x8*)(base + kb);
                    pa.h[1] = *(const bf16x8*)(base + 16 + kb);
                }
                AFrag vb;
                {
                    int dv = ntv * 16 + n0;
                    int kb = ks * 32 + (lane >> 4) * 8;
                    const __bf16* base = Vc + dv * 64 + kb;
                    vb.h[0] = *(const bf16x8*)(base);
                    vb.h[1] = *(const bf16x8*)(base + 16);
                }
                acc[ntv] = WMMA_BF16(pa.v, vb.v, acc[ntv]);
            }
        }
        __syncthreads();
    }

    // ---- normalize and store heads [B, S, H*DV] --------------------------
#pragma unroll
    for (int r = 0; r < 8; ++r) {
        float inv = 1.0f / fmaxf(l_i[r], 1.0e-20f);
        int q = q0 + w * 16 + m0 + r;
        size_t row = ((size_t)b * S_ + q) * (H_ * DV_) + (size_t)h * DV_;
#pragma unroll
        for (int ntv = 0; ntv < 4; ++ntv)
            O[row + ntv * 16 + n0] = (__bf16)(acc[ntv][r] * inv);
    }
}

// ---------------------------------------------------------------------------
// Kernel 3: output projection.  heads [B*S, 1024] @ W0 [1024, 1024] -> f32.
// W0 pre-transposed: w0t [1024(n=d), 1024(k=c)].  k-step 64, double-buffered.
// grid = (B*S/128, 1024/64), block = 256 (8 waves).
// ---------------------------------------------------------------------------
__global__ void __launch_bounds__(256)
out_proj(const __bf16* __restrict__ Hd,
         const __bf16* __restrict__ W0t,
         float* __restrict__ out)
{
    __shared__ __bf16 Xs[2][128 * 64];
    __shared__ __bf16 Ws[2][64 * 64];   // [n][k]

    const int t0 = blockIdx.x * 128;
    const int c0 = blockIdx.y * 64;

    const int tid  = threadIdx.x;
    const int lane = tid & 31;
    const int wid  = tid >> 5;

    f32x8 acc[4] = {};
    const __bf16* htile0 = Hd + (size_t)t0 * 1024;
    const __bf16* wtile0 = W0t + (size_t)c0 * 1024;

#if defined(HAVE_TDM)
    if (tid < 32) {
        tdm_load_2d(lds_offset(Xs[0]), htile0, 64, 128, 1024ull);
        tdm_load_2d(lds_offset(Ws[0]), wtile0, 64, 64,  1024ull);
    }
#endif

    for (int k0 = 0; k0 < 1024; k0 += 64) {
        const int cur = (k0 >> 6) & 1;
#if defined(HAVE_TDM)
        if (tid < 32) TDM_WAIT();
        __syncthreads();
        if (k0 + 64 < 1024 && tid < 32) {
            tdm_load_2d(lds_offset(Xs[cur ^ 1]), htile0 + k0 + 64, 64, 128, 1024ull);
            tdm_load_2d(lds_offset(Ws[cur ^ 1]), wtile0 + k0 + 64, 64, 64,  1024ull);
        }
#else
        {
            const uint4* xg4 = (const uint4*)(htile0 + k0);
            uint4* xs4 = (uint4*)Xs[cur];
#pragma unroll
            for (int t = 0; t < 4; ++t) {
                int i = tid + t * 256;
                int r = i >> 3, c = i & 7;
                xs4[r * 8 + c] = xg4[(size_t)r * 128 + c];
            }
            const uint4* wg4 = (const uint4*)(wtile0 + k0);
            uint4* ws4 = (uint4*)Ws[cur];
#pragma unroll
            for (int t = 0; t < 2; ++t) {
                int i = tid + t * 256;
                int r = i >> 3, c = i & 7;
                ws4[r * 8 + c] = wg4[(size_t)r * 128 + c];
            }
        }
        __syncthreads();
#endif
        const __bf16* Xc = Xs[cur];
        const __bf16* Wc = Ws[cur];
#pragma unroll
        for (int ks = 0; ks < 2; ++ks) {
            AFrag a;
            {
                int m  = lane & 15;
                int kb = (lane >> 4) * 8;
                const __bf16* base = Xc + (wid * 16 + m) * 64 + ks * 32;
                a.h[0] = *(const bf16x8*)(base + kb);
                a.h[1] = *(const bf16x8*)(base + 16 + kb);
            }
#pragma unroll
            for (int nt = 0; nt < 4; ++nt) {
                AFrag bf;
                int n  = nt * 16 + (lane & 15);
                int kb = (lane >> 4) * 8;
                const __bf16* base = Wc + n * 64 + ks * 32 + kb;
                bf.h[0] = *(const bf16x8*)(base);
                bf.h[1] = *(const bf16x8*)(base + 16);
                acc[nt] = WMMA_BF16(a.v, bf.v, acc[nt]);
            }
        }
        __syncthreads();
    }

    const int n0 = lane & 15;
    const int m0 = (lane >> 4) * 8;
#pragma unroll
    for (int nt = 0; nt < 4; ++nt)
#pragma unroll
        for (int r = 0; r < 8; ++r) {
            int t = t0 + wid * 16 + m0 + r;
            out[(size_t)t * 1024 + c0 + nt * 16 + n0] = acc[nt][r];
        }
}

// ---------------------------------------------------------------------------
// Launch
// ---------------------------------------------------------------------------
extern "C" void kernel_launch(void* const* d_in, const int* in_sizes, int n_in,
                              void* d_out, int out_size, void* d_ws, size_t ws_size,
                              hipStream_t stream) {
    const float* x  = (const float*)d_in[0];   // [B,S,D]
    const float* Wq = (const float*)d_in[1];   // [H,D,DK]
    const float* Wk = (const float*)d_in[2];   // [H,D,DK]
    const float* Wv = (const float*)d_in[3];   // [H,D,DV]
    const float* W0 = (const float*)d_in[4];   // [H*DV, D]

    const size_t nX  = (size_t)B_ * S_ * D_;
    const size_t nW  = (size_t)H_ * D_ * DK_;
    const size_t nW0 = (size_t)H_ * DV_ * D_;
    const size_t nQ  = (size_t)B_ * H_ * S_ * DK_;
    const size_t nHd = (size_t)B_ * S_ * H_ * DV_;

    char* p = (char*)d_ws;
    __bf16* xb  = (__bf16*)p;  p += nX  * 2;
    __bf16* wqt = (__bf16*)p;  p += nW  * 2;   // transposed [H, DK, D]
    __bf16* wkt = (__bf16*)p;  p += nW  * 2;
    __bf16* wvt = (__bf16*)p;  p += nW  * 2;
    __bf16* w0t = (__bf16*)p;  p += nW0 * 2;   // transposed [D, H*DV]
    __bf16* Qb  = (__bf16*)p;  p += nQ  * 2;
    __bf16* Kb  = (__bf16*)p;  p += nQ  * 2;
    __bf16* Vtb = (__bf16*)p;  p += nQ  * 2;   // transposed [B,H,DV,S]
    __bf16* Hdb = (__bf16*)p;  p += nHd * 2;

    cvt_f32_bf16<<<(int)((nX + 255) / 256), 256, 0, stream>>>(x, xb, (int)nX);

    // weights: convert + transpose so every GEMM tile is row-major for TDM
    cvt_transpose_bf16<<<dim3((D_ * DK_ + 255) / 256, 1, H_), 256, 0, stream>>>(Wq, wqt, D_, DK_);
    cvt_transpose_bf16<<<dim3((D_ * DK_ + 255) / 256, 1, H_), 256, 0, stream>>>(Wk, wkt, D_, DK_);
    cvt_transpose_bf16<<<dim3((D_ * DK_ + 255) / 256, 1, H_), 256, 0, stream>>>(Wv, wvt, D_, DK_);
    cvt_transpose_bf16<<<dim3((1024 * 1024 + 255) / 256, 1, 1), 256, 0, stream>>>(W0, w0t, 1024, 1024);

    qkv_gemm<<<dim3(S_ / 128, 3 * H_, B_), 256, 0, stream>>>(xb, wqt, wkt, wvt, Qb, Kb, Vtb);

    flash_attn<<<dim3(S_ / 64, H_, B_), 128, 0, stream>>>(Qb, Kb, Vtb, Hdb);

    out_proj<<<dim3((B_ * S_) / 128, 1024 / 64), 256, 0, stream>>>(Hdb, w0t, (float*)d_out);
}
